// CostVRefinementBlock_48421461295724
// MI455X (gfx1250) — matrix operand order repaired and checked by reading, not compiled, not run on hardware
//
#include <hip/hip_runtime.h>
#include <hip/hip_bf16.h>

typedef _Float16 f16;
typedef __attribute__((ext_vector_type(16))) _Float16     v16h;
typedef __attribute__((ext_vector_type(8)))  float        v8f;
typedef unsigned int u32;
typedef __attribute__((ext_vector_type(2))) unsigned int  u32x2;
typedef __attribute__((ext_vector_type(4))) unsigned int  u32x4;
typedef __attribute__((ext_vector_type(4))) int           i32x4;
typedef __attribute__((ext_vector_type(8))) int           i32x8;

#ifndef __has_builtin
#define __has_builtin(x) 0
#endif
#if __has_builtin(__builtin_amdgcn_tensor_load_to_lds) && \
    __has_builtin(__builtin_amdgcn_s_wait_tensorcnt)
#define HAVE_TDM 1
#else
#define HAVE_TDM 0
#endif

#define NVOX (48*48*48)          // 110592
#define PW 52
#define PH 50
#define PD 50
#define PSTRD (PH*PW)            // 2600
#define PSTRC ((size_t)PD*PH*PW) // 130000
#define CTOT 352

// workspace layout (bytes)
#define ACT_BYTES ((size_t)CTOT*PSTRC*2)            // 91,520,000
#define WARP_OFF  ACT_BYTES
#define WARP_BYTES ((size_t)16*NVOX*4)              //  7,077,888
#define PACKW_OFF (WARP_OFF + WARP_BYTES)
#define PACKW_HALVES 1050624                        //  2,101,248 B
#define STATS_OFF (PACKW_OFF + (size_t)PACKW_HALVES*2)

// ---------------------------------------------------------------------------
// Weight packing: f32 [O][I][3][3][3] -> f16 A-fragments (ISA 7.12.2 16x32 f16
// layout), K zero-padded per 32-chunk so ragged channel tails contribute 0.
__global__ void k_packw(const float* __restrict__ w, f16* __restrict__ out,
                        int Cin, int nkc, int npacked) {
  int p = blockIdx.x * blockDim.x + threadIdx.x;
  if (p >= npacked) return;
  int e    = p & 15;
  int lane = (p >> 4) & 31;
  int tile = p >> 9;
  int kc   = tile % nkc;
  int t    = (tile / nkc) % 27;
  int cot  = tile / (nkc * 27);
  int m  = lane & 15;
  int k  = e + ((e < 8) ? 0 : 8) + ((lane >= 16) ? 8 : 0);
  int ci = kc * 32 + k;
  int co = cot * 16 + m;
  float v = (ci < Cin) ? w[((size_t)co * Cin + ci) * 27 + t] : 0.f;
  out[p] = (f16)v;
}

// ---------------------------------------------------------------------------
__global__ void k_upsample(const float* __restrict__ ctx, f16* __restrict__ act) {
  int idx = blockIdx.x * blockDim.x + threadIdx.x;
  if (idx >= NVOX) return;
  int x = idx % 48, y = (idx / 48) % 48, z = idx / 2304;
  float sz = 0.5f * z - 0.25f, sy = 0.5f * y - 0.25f, sx = 0.5f * x - 0.25f;
  float fz = floorf(sz), fy = floorf(sy), fx = floorf(sx);
  float wz = sz - fz, wy = sy - fy, wx = sx - fx;
  int z0 = min(max((int)fz, 0), 23), z1 = min(max((int)fz + 1, 0), 23);
  int y0 = min(max((int)fy, 0), 23), y1 = min(max((int)fy + 1, 0), 23);
  int x0 = min(max((int)fx, 0), 23), x1 = min(max((int)fx + 1, 0), 23);
  size_t o = (size_t)(z + 1) * PSTRD + (size_t)(y + 1) * PW + (x + 1);
  for (int c = 0; c < 32; ++c) {
    const float* v = ctx + (size_t)c * 13824;
    float r =
      ((v[(z0*24+y0)*24+x0]*(1.f-wx) + v[(z0*24+y0)*24+x1]*wx)*(1.f-wy) +
       (v[(z0*24+y1)*24+x0]*(1.f-wx) + v[(z0*24+y1)*24+x1]*wx)*wy)*(1.f-wz) +
      ((v[(z1*24+y0)*24+x0]*(1.f-wx) + v[(z1*24+y0)*24+x1]*wx)*(1.f-wy) +
       (v[(z1*24+y1)*24+x0]*(1.f-wx) + v[(z1*24+y1)*24+x1]*wx)*wy)*wz;
    act[(size_t)c * PSTRC + o] = (f16)r;
  }
}

// ---------------------------------------------------------------------------
__global__ void k_warp(const float* __restrict__ mov, const float* __restrict__ flw,
                       float* __restrict__ wrp) {
  int idx = blockIdx.x * blockDim.x + threadIdx.x;
  if (idx >= NVOX) return;
  int x = idx % 48, y = (idx / 48) % 48, z = idx / 2304;
  float az = (float)z + flw[idx];
  float ay = (float)y + flw[NVOX + idx];
  float ax = (float)x + flw[2 * NVOX + idx];
  float fz = floorf(az), fy = floorf(ay), fx = floorf(ax);
  float wz = az - fz, wy = ay - fy, wx = ax - fx;
  int z0 = min(max((int)fz, 0), 47), z1 = min(max((int)fz + 1, 0), 47);
  int y0 = min(max((int)fy, 0), 47), y1 = min(max((int)fy + 1, 0), 47);
  int x0 = min(max((int)fx, 0), 47), x1 = min(max((int)fx + 1, 0), 47);
  size_t o00 = (size_t)z0 * 2304, o01 = (size_t)z1 * 2304;
  size_t a = o00 + (size_t)y0 * 48, b = o00 + (size_t)y1 * 48;
  size_t c2 = o01 + (size_t)y0 * 48, d = o01 + (size_t)y1 * 48;
  for (int c = 0; c < 16; ++c) {
    const float* v = mov + (size_t)c * NVOX;
    float r =
      ((v[a+x0]*(1.f-wx) + v[a+x1]*wx)*(1.f-wy) +
       (v[b+x0]*(1.f-wx) + v[b+x1]*wx)*wy)*(1.f-wz) +
      ((v[c2+x0]*(1.f-wx) + v[c2+x1]*wx)*(1.f-wy) +
       (v[d+x0]*(1.f-wx) + v[d+x1]*wx)*wy)*wz;
    wrp[(size_t)c * NVOX + idx] = r;
  }
}

// ---------------------------------------------------------------------------
__global__ void k_stats(const float* __restrict__ wrp, const float* __restrict__ fx,
                        float* __restrict__ stats) {
  float sw = 0.f, sw2 = 0.f, sf = 0.f, sf2 = 0.f;
  int N = 16 * NVOX;
  for (int i = blockIdx.x * blockDim.x + threadIdx.x; i < N;
       i += gridDim.x * blockDim.x) {
    float a = wrp[i]; sw += a; sw2 += a * a;
    float b = fx[i];  sf += b; sf2 += b * b;
  }
  for (int off = 16; off; off >>= 1) {
    sw  += __shfl_down(sw, off);
    sw2 += __shfl_down(sw2, off);
    sf  += __shfl_down(sf, off);
    sf2 += __shfl_down(sf2, off);
  }
  if ((threadIdx.x & 31) == 0) {
    atomicAdd(&stats[0], sw);  atomicAdd(&stats[1], sw2);
    atomicAdd(&stats[2], sf);  atomicAdd(&stats[3], sf2);
  }
}

__global__ void k_finalize(float* __restrict__ stats) {
  float n  = 16.f * (float)NVOX;
  float mw = stats[0] / n, mf = stats[2] / n;
  float vw = (stats[1] - n * mw * mw) / (n - 1.f);
  float vf = (stats[3] - n * mf * mf) / (n - 1.f);
  stats[4] = 0.5f * (mw + mf);
  stats[5] = 0.5f * (sqrtf(fmaxf(vw, 0.f)) + sqrtf(fmaxf(vf, 0.f)));
}

// ---------------------------------------------------------------------------
__global__ void k_packff(const float* __restrict__ flw, const float* __restrict__ fx,
                         f16* __restrict__ act) {
  int idx = blockIdx.x * blockDim.x + threadIdx.x;
  if (idx >= NVOX) return;
  int x = idx % 48, y = (idx / 48) % 48, z = idx / 2304;
  size_t o = (size_t)(z + 1) * PSTRD + (size_t)(y + 1) * PW + (x + 1);
  for (int c = 0; c < 3; ++c)
    act[(size_t)(32 + c) * PSTRC + o] = (f16)flw[(size_t)c * NVOX + idx];
  for (int c = 0; c < 16; ++c)
    act[(size_t)(160 + c) * PSTRC + o] = (f16)fx[(size_t)c * NVOX + idx];
}

// ---------------------------------------------------------------------------
__global__ void k_cv(const float* __restrict__ fx, const float* __restrict__ wrp,
                     const float* __restrict__ stats, f16* __restrict__ act) {
  int idx = blockIdx.x * blockDim.x + threadIdx.x;
  if (idx >= NVOX) return;
  int x = idx % 48, y = (idx / 48) % 48, z = idx / 2304;
  float m = stats[4];
  float inv = 1.f / stats[5];
  float fn[16];
#pragma unroll
  for (int c = 0; c < 16; ++c) fn[c] = (fx[(size_t)c * NVOX + idx] - m) * inv;
  size_t o = (size_t)(z + 1) * PSTRD + (size_t)(y + 1) * PW + (x + 1);
  int sidx = 0;
  for (int i = -2; i <= 2; ++i)
    for (int j = -2; j <= 2; ++j)
      for (int k = -2; k <= 2; ++k) {
        int zz = z + i, yy = y + j, xx = x + k;
        float acc = 0.f;
        if ((unsigned)zz < 48u && (unsigned)yy < 48u && (unsigned)xx < 48u) {
          size_t wi = (size_t)zz * 2304 + (size_t)yy * 48 + xx;
#pragma unroll
          for (int c = 0; c < 16; ++c)
            acc += fn[c] * ((wrp[(size_t)c * NVOX + wi] - m) * inv);
        }
        acc *= (1.f / 16.f);
        acc = acc > 0.f ? acc : 0.05f * acc;
        act[(size_t)(35 + sidx) * PSTRC + o] = (f16)acc;
        ++sidx;
      }
}

// ---------------------------------------------------------------------------
#if HAVE_TDM
// Issue a TDM load of one K-chunk's weight slab (27 taps x 512 halves, global
// row stride nkc*512 halves) into LDS.  D# per CDNA5 ISA ch.8.
__device__ __forceinline__ void tdm_load_weights(const f16* gsrc, f16* ldst,
                                                 int nkc) {
  unsigned long long ga = (unsigned long long)(uintptr_t)gsrc;
  unsigned lds = (unsigned)(uintptr_t)ldst;
  u32x4 g0 = { 1u,                                    // count=1 (valid user D#)
               lds,                                   // lds_addr
               (unsigned)(ga & 0xFFFFFFFFu),          // global_addr[31:0]
               (unsigned)((ga >> 32) & 0x01FFFFFFu) | 0x80000000u }; // type=2
  i32x8 g1 = { 0x00010000,            // data_size=1 (2B), no mask/pad/iterate
               (int)(512u << 16),     // tensor_dim0 = 512 (lo16 -> [31:16])
               (int)(27u << 16),      // tensor_dim0 hi=0 | tensor_dim1=27
               (int)(512u << 16),     // tensor_dim1 hi=0 | tile_dim0=512
               27,                    // tile_dim1=27, tile_dim2=0
               nkc * 512,             // tensor_dim0_stride[31:0]
               0, 0 };                // stride0 hi / stride1
  i32x4 g2 = { 1, 1, 0, 0 };          // tensor_dim2=1, tensor_dim3=1
  i32x4 g3 = { 0, 0, 0, 0 };
#if __clang_major__ >= 23
  i32x8 g4 = { 0, 0, 0, 0, 0, 0, 0, 0 };
  __builtin_amdgcn_tensor_load_to_lds(g0, g1, g2, g3, g4, 0);
#else
  __builtin_amdgcn_tensor_load_to_lds(g0, g1, g2, g3, 0);
#endif
}
#endif

// ---------------------------------------------------------------------------
// Implicit-GEMM 3x3x3 conv via v_wmma_f32_16x16x32_f16.
// One wave per (16 out-ch) x (48-wide x-row); 3 N-tiles share each A fragment.
// Weights stream into LDS via the Tensor Data Mover (fallback: global loads).
__global__ __launch_bounds__(32) void k_conv(
    f16* __restrict__ act, const f16* __restrict__ packW,
    const float* __restrict__ bias, float* __restrict__ dout,
    int nkc, int chOut) {
  __shared__ __align__(16) f16 ldsB[9 * 52 * 32];   // [rr][x 0..49(+2)][ci]
#if HAVE_TDM
  __shared__ __align__(16) f16 wbuf[27 * 512];      // one K-chunk of A-frags
#endif
  int lane = threadIdx.x;
  int row = blockIdx.x;                             // 0..2303 = (z,y)
  int cot = blockIdx.y;
  int y = row % 48, z = row / 48;
  int hi = lane >> 4, n = lane & 15, kb = hi << 3;
  v8f acc[3] = {};
  for (int kc = 0; kc < nkc; ++kc) {
#if HAVE_TDM
    // async DMA of this chunk's weight slab; overlaps manual act staging
    tdm_load_weights(packW + ((size_t)cot * 27 * nkc + kc) * 512, wbuf, nkc);
#endif
    // ---- stage 32ci x 3z x 3y x 50x act slab into LDS, [rr][x][ci] ----
    int ciG = kc * 32 + lane;   // ci >= Cin reads are nulled by zero weights
    const f16* pz = act + (size_t)ciG * PSTRC + (size_t)z * PSTRD +
                    (size_t)y * PW;
    for (int zz = 0; zz < 3; ++zz, pz += PSTRD) {
      const f16* py = pz;
      for (int yy = 0; yy < 3; ++yy, py += PW) {
        union { u32x2 q[13]; f16 h[52]; } ru;
        const u32x2* gq = (const u32x2*)py;          // 8B aligned by layout
#pragma unroll
        for (int q = 0; q < 12; ++q) ru.q[q] = gq[q];
        ru.q[12].x = *(const u32*)(py + 48);
        f16* ldst = &ldsB[(size_t)(zz * 3 + yy) * 52 * 32 + lane];
#pragma unroll
        for (int xi = 0; xi < 50; ++xi) ldst[xi * 32] = ru.h[xi];
      }
    }
#if HAVE_TDM
    __builtin_amdgcn_s_wait_tensorcnt(0);
    asm volatile("" ::: "memory");
#endif
    __syncthreads();
    // ---- 27 taps x 3 N-tiles, fully unrolled: all-immediate LDS offsets ----
#pragma unroll
    for (int rr = 0; rr < 9; ++rr) {
#pragma unroll
      for (int dxp = 0; dxp < 3; ++dxp) {
        const int t = rr * 3 + dxp;
        union { u32x4 u[2]; v16h v; } A;
#if HAVE_TDM
        const u32x4* ap = (const u32x4*)&wbuf[(t << 9) + (lane << 4)];
#else
        const u32x4* ap = (const u32x4*)(packW +
            ((((size_t)cot * 27 + t) * nkc + kc) << 9) + ((size_t)lane << 4));
        __builtin_prefetch((const void*)(ap + 64), 0, 0);
#endif
        A.u[0] = ap[0]; A.u[1] = ap[1];
#pragma unroll
        for (int tl = 0; tl < 3; ++tl) {
          union { u32x4 u[2]; v16h v; } B;
          const u32x4* bp =
              (const u32x4*)&ldsB[((rr * 52 + dxp + (tl << 4) + n) << 5) + kb];
          B.u[0] = bp[0]; B.u[1] = bp[2];
          acc[tl] = __builtin_amdgcn_wmma_f32_16x16x32_f16(
              false, A.v, false, B.v, (short)0, acc[tl], false, false);
        }
      }
    }
    __syncthreads();
  }
  // ---- epilogue: bias + leaky(0.02), f16 store (and f32 for last layer) ----
  int coBase = cot << 4;
#pragma unroll
  for (int tl = 0; tl < 3; ++tl) {
    size_t ob = (size_t)(z + 1) * PSTRD + (size_t)(y + 1) * PW +
                ((tl << 4) + 1 + n);
#pragma unroll
    for (int r = 0; r < 8; ++r) {
      int m = r + (hi << 3);
      float v = acc[tl][r] + bias[coBase + m];
      v = v > 0.f ? v : 0.02f * v;
      act[(size_t)(chOut + coBase + m) * PSTRC + ob] = (f16)v;
      if (dout)
        dout[(size_t)(coBase + m) * NVOX + (size_t)z * 2304 + (size_t)y * 48 +
             (tl << 4) + n] = v;
    }
  }
}

// ---------------------------------------------------------------------------
__global__ void k_refine(const f16* __restrict__ act, const float* __restrict__ w4,
                         const float* __restrict__ b4, const float* __restrict__ flw,
                         float* __restrict__ dout) {
  __shared__ float wl[1296];
  __shared__ float bl[3];
  for (int i = threadIdx.x; i < 1296; i += blockDim.x) wl[i] = w4[i];
  if (threadIdx.x < 3) bl[threadIdx.x] = b4[threadIdx.x];
  __syncthreads();
  int idx = blockIdx.x * blockDim.x + threadIdx.x;
  if (idx >= NVOX) return;
  int x = idx % 48, y = (idx / 48) % 48, z = idx / 2304;
  float a0 = bl[0], a1 = bl[1], a2 = bl[2];
  for (int i = 0; i < 16; ++i) {
    const f16* v = act + (size_t)(320 + i) * PSTRC + (size_t)z * PSTRD +
                   (size_t)y * PW + x;
    int t = 0;
#pragma unroll
    for (int kd = 0; kd < 3; ++kd)
#pragma unroll
      for (int kh = 0; kh < 3; ++kh)
#pragma unroll
        for (int kw = 0; kw < 3; ++kw) {
          float s = (float)v[(size_t)kd * PSTRD + kh * PW + kw];
          a0 += wl[(0 * 16 + i) * 27 + t] * s;
          a1 += wl[(1 * 16 + i) * 27 + t] * s;
          a2 += wl[(2 * 16 + i) * 27 + t] * s;
          ++t;
        }
  }
  dout[(size_t)16 * NVOX + idx] = flw[idx] + a0;
  dout[(size_t)17 * NVOX + idx] = flw[NVOX + idx] + a1;
  dout[(size_t)18 * NVOX + idx] = flw[2 * NVOX + idx] + a2;
}

// ---------------------------------------------------------------------------
extern "C" void kernel_launch(void* const* d_in, const int* in_sizes, int n_in,
                              void* d_out, int out_size, void* d_ws, size_t ws_size,
                              hipStream_t stream) {
  const float* mov = (const float*)d_in[0];
  const float* fx  = (const float*)d_in[1];
  const float* flw = (const float*)d_in[2];
  const float* ctx = (const float*)d_in[3];
  const float* w0 = (const float*)d_in[4];   const float* b0 = (const float*)d_in[5];
  const float* w1 = (const float*)d_in[6];   const float* b1 = (const float*)d_in[7];
  const float* w2 = (const float*)d_in[8];   const float* b2 = (const float*)d_in[9];
  const float* w3 = (const float*)d_in[10];  const float* b3 = (const float*)d_in[11];
  const float* w4 = (const float*)d_in[12];  const float* b4 = (const float*)d_in[13];
  float* dout = (float*)d_out;
  char* ws = (char*)d_ws;
  f16*   act   = (f16*)ws;
  float* wrp   = (float*)(ws + WARP_OFF);
  f16*   pw    = (f16*)(ws + PACKW_OFF);
  float* stats = (float*)(ws + STATS_OFF);

  hipMemsetAsync(act, 0, ACT_BYTES, stream);          // zero halos
  hipMemsetAsync(stats, 0, 6 * sizeof(float), stream);

  { int np = 4 * 27 * 6  * 512; k_packw<<<(np + 255) / 256, 256, 0, stream>>>(w0, pw + 0,      176, 6,  np); }
  { int np = 3 * 27 * 8  * 512; k_packw<<<(np + 255) / 256, 256, 0, stream>>>(w1, pw + 331776, 240, 8,  np); }
  { int np = 2 * 27 * 9  * 512; k_packw<<<(np + 255) / 256, 256, 0, stream>>>(w2, pw + 663552, 288, 9,  np); }
  { int np = 1 * 27 * 10 * 512; k_packw<<<(np + 255) / 256, 256, 0, stream>>>(w3, pw + 912384, 320, 10, np); }

  int blocks = (NVOX + 255) / 256;
  k_upsample<<<blocks, 256, 0, stream>>>(ctx, act);
  k_warp    <<<blocks, 256, 0, stream>>>(mov, flw, wrp);
  k_stats   <<<256, 256, 0, stream>>>(wrp, fx, stats);
  k_finalize<<<1, 1, 0, stream>>>(stats);
  k_packff  <<<blocks, 256, 0, stream>>>(flw, fx, act);
  k_cv      <<<blocks, 256, 0, stream>>>(fx, wrp, stats, act);

  dim3 g0(2304, 4), g1(2304, 3), g2(2304, 2), g3(2304, 1);
  k_conv<<<g0, 32, 0, stream>>>(act, pw + 0,      b0, nullptr, 6,  176);
  k_conv<<<g1, 32, 0, stream>>>(act, pw + 331776, b1, nullptr, 8,  240);
  k_conv<<<g2, 32, 0, stream>>>(act, pw + 663552, b2, nullptr, 9,  288);
  k_conv<<<g3, 32, 0, stream>>>(act, pw + 912384, b3, dout,    10, 320);

  k_refine<<<blocks, 256, 0, stream>>>(act, w4, b4, flw, dout);
}
